// PointNetSetAbstractionMsg_41111426957682
// MI455X (gfx1250) — matrix operand order repaired and checked by reading, not compiled, not run on hardware
//
#include <hip/hip_runtime.h>
#include <hip/hip_bf16.h>

typedef __attribute__((ext_vector_type(16))) _Float16 v16h;
typedef __attribute__((ext_vector_type(8)))  _Float16 v8h;
typedef __attribute__((ext_vector_type(8)))  float    v8f;

#define NPOINT 256
#define NBATCH 8            // b*t
#define NPTS   1024
#define BN_EPS 1e-5f

// ---------------------------------------------------------------------------
// 1) Farthest point sampling: one block per batch, 256 threads, 4 pts/thread.
//    Matches jax scan semantics: sampled[0]=0, then argmax of running min-dist
//    (ties -> smallest index, like jnp.argmax).
// ---------------------------------------------------------------------------
__global__ void fps_kernel(const float* __restrict__ xyz, int* __restrict__ fps_idx,
                           float* __restrict__ newxyz, float* __restrict__ out_newxyz) {
    const int B8 = blockIdx.x, tid = threadIdx.x;
    const float* xb = xyz + (size_t)B8 * 3 * NPTS;
    float px[4], py[4], pz[4], dist[4];
#pragma unroll
    for (int i = 0; i < 4; ++i) {
        int n = tid + i * 256;
        px[i] = xb[n]; py[i] = xb[NPTS + n]; pz[i] = xb[2 * NPTS + n];
        dist[i] = 1e10f;
    }
    __shared__ float sD[256];
    __shared__ int   sI[256];
    __shared__ float sC[3];
    __shared__ int   sFar;
    if (tid == 0) sFar = 0;
    __syncthreads();
    for (int s = 0; s < NPOINT; ++s) {
        int far = sFar;
        if (tid == (far & 255)) {            // owner broadcasts centroid coords
            int slot = far >> 8;
            sC[0] = px[slot]; sC[1] = py[slot]; sC[2] = pz[slot];
        }
        __syncthreads();
        float cx = sC[0], cy = sC[1], cz = sC[2];
        if (tid == 0) fps_idx[B8 * NPOINT + s] = far;
        if (tid < 3) {
            float v = sC[tid];
            newxyz[((size_t)B8 * NPOINT + s) * 3 + tid] = v;
            out_newxyz[((size_t)B8 * 3 + tid) * NPOINT + s] = v;   // (b,t,3,S)
        }
        float bd = -1e30f; int bi = 0;
#pragma unroll
        for (int i = 0; i < 4; ++i) {
            float dx = px[i] - cx, dy = py[i] - cy, dz = pz[i] - cz;
            float d = dx * dx + dy * dy + dz * dz;
            if (d < dist[i]) dist[i] = d;
            if (dist[i] > bd) { bd = dist[i]; bi = tid + i * 256; }
        }
        sD[tid] = bd; sI[tid] = bi;
        __syncthreads();
        for (int off = 128; off > 0; off >>= 1) {
            if (tid < off) {
                float dO = sD[tid + off]; int iO = sI[tid + off];
                if (dO > sD[tid] || (dO == sD[tid] && iO < sI[tid])) { sD[tid] = dO; sI[tid] = iO; }
            }
            __syncthreads();
        }
        if (tid == 0) sFar = sI[0];
        __syncthreads();
    }
}

// ---------------------------------------------------------------------------
// 2) Ball query + gather: one block per (batch, centroid). Index-ordered
//    selection via an LDS bitmask scan (== reference's sort-with-sentinel),
//    pad with first index, write f16 feature rows X[p][ICpad=160].
// ---------------------------------------------------------------------------
__global__ void group_kernel(const float* __restrict__ xyz, const float* __restrict__ pts,
                             const float* __restrict__ newxyz, float r2, int K, int ICpad,
                             _Float16* __restrict__ X) {
    const int g = blockIdx.x, tid = threadIdx.x;     // 128 threads
    const int B8 = g >> 8;
    const float* xb = xyz + (size_t)B8 * 3 * NPTS;
    const float  nx = newxyz[(size_t)g * 3 + 0];
    const float  ny = newxyz[(size_t)g * 3 + 1];
    const float  nz = newxyz[(size_t)g * 3 + 2];
    __shared__ int      sgidx[64];
    __shared__ unsigned sflags[32];
    __shared__ int      scnt;
    if (tid < 32) sflags[tid] = 0u;
    __syncthreads();
    for (int n = tid * 8, e = tid * 8 + 8; n < e; ++n) {
        float dx = xb[n] - nx, dy = xb[NPTS + n] - ny, dz = xb[2 * NPTS + n] - nz;
        if (dx * dx + dy * dy + dz * dz <= r2) atomicOr(&sflags[n >> 5], 1u << (n & 31));
    }
    __syncthreads();
    if (tid == 0) {
        int cnt = 0;
        for (int w = 0; w < 32 && cnt < K; ++w) {
            unsigned m = sflags[w];
            while (m && cnt < K) { int b = __ffs(m) - 1; sgidx[cnt++] = w * 32 + b; m &= m - 1; }
        }
        if (cnt == 0) { sgidx[0] = 0; cnt = 1; }
        scnt = cnt;
    }
    __syncthreads();
    if (tid >= scnt && tid < K) sgidx[tid] = sgidx[0];
    __syncthreads();
    const float* pb = pts + (size_t)B8 * 128 * NPTS;
    for (int k = 0; k < K; ++k) {
        int idx = sgidx[k];
        _Float16* xr = X + ((size_t)g * K + k) * ICpad;
        for (int c = tid; c < ICpad; c += blockDim.x) {
            float v;
            if (c < 128)      v = pb[(size_t)c * NPTS + idx];
            else if (c == 128) v = xb[idx] - nx;
            else if (c == 129) v = xb[NPTS + idx] - ny;
            else if (c == 130) v = xb[2 * NPTS + idx] - nz;
            else              v = 0.0f;
            xr[c] = (_Float16)v;
        }
    }
}

// ---------------------------------------------------------------------------
// 3) Weight conversion f32 -> padded f16
// ---------------------------------------------------------------------------
__global__ void wconvert_kernel(const float* __restrict__ W, _Float16* __restrict__ Wt,
                                int OC, int IC, int ICpad) {
    int i = blockIdx.x * blockDim.x + threadIdx.x;
    if (i >= OC * ICpad) return;
    int oc = i / ICpad, c = i % ICpad;
    Wt[i] = (_Float16)((c < IC) ? W[oc * IC + c] : 0.0f);
}

// ---------------------------------------------------------------------------
// 4) WMMA GEMM: Y[p][oc] = sum_c X[p][c]*W[oc][c] + bias[oc], f16 out.
//    One wave per 16x16 tile, fragments use contiguous k order (k = 16*hi + s) in A and B.
//    blockDim=128 (4 waves -> 4 oc-tiles), grid=(P/16, OC/64).
// ---------------------------------------------------------------------------
__global__ void gemm_bias_kernel(const _Float16* __restrict__ X, const _Float16* __restrict__ Wt,
                                 const float* __restrict__ bias, _Float16* __restrict__ Y,
                                 int P, int OC, int ICpad) {
    const int lane = threadIdx.x & 31;
    const int wave = threadIdx.x >> 5;
    const int postile = blockIdx.x;
    const int octile  = blockIdx.y * 4 + wave;
    const int rc  = lane & 15;          // A-row / B-col
    const int hi  = lane >> 4;          // upper half-wave selects the other K-halves
    const _Float16* wrow = Wt + (size_t)(octile * 16 + rc) * ICpad;
    const _Float16* xrow = X  + (size_t)(postile * 16 + rc) * ICpad;
    v8f acc = {};
    for (int kb = 0; kb < ICpad; kb += 32) {
        __builtin_prefetch(xrow + kb + 32, 0, 1);           // global_prefetch_b8
        // A (16x32 f16): lanes<16 hold K 0..15, lanes>=16 hold K 16..31 (contiguous)
        v16h a = *(const v16h*)(wrow + kb + (hi ? 16 : 0));
        // B (32x16 f16): lanes<16 hold K 0..15, lanes>=16 hold K 16..31 (contiguous)
        v16h b = *(const v16h*)(xrow + kb + (hi ? 16 : 0));
        acc = __builtin_amdgcn_wmma_f32_16x16x32_f16(false, a, false, b,
                                                     (short)0, acc, false, false);
    }
    // C/D layout: element r -> oc = octile*16 + r + (hi?8:0), col = rc
    const int ocbase = octile * 16 + (hi ? 8 : 0);
    v8h outv;
#pragma unroll
    for (int r = 0; r < 8; ++r) outv[r] = (_Float16)(acc[r] + bias[ocbase + r]);
    *(v8h*)(Y + (size_t)(postile * 16 + rc) * OC + ocbase) = outv;
}

// ---------------------------------------------------------------------------
// 5) Per-channel BN statistics (one block per channel; no atomics needed)
// ---------------------------------------------------------------------------
__global__ void stats_kernel(const _Float16* __restrict__ Y, int P, int OC,
                             float* __restrict__ stats) {
    const int ch = blockIdx.x, tid = threadIdx.x;
    float s = 0.f, s2 = 0.f;
    for (int p = tid; p < P; p += blockDim.x) {
        float v = (float)Y[(size_t)p * OC + ch];
        s += v; s2 += v * v;
    }
    __shared__ float shs[256], shq[256];
    shs[tid] = s; shq[tid] = s2;
    __syncthreads();
    for (int off = 128; off > 0; off >>= 1) {
        if (tid < off) { shs[tid] += shs[tid + off]; shq[tid] += shq[tid + off]; }
        __syncthreads();
    }
    if (tid == 0) { stats[ch] = shs[0]; stats[256 + ch] = shq[0]; }
}

// ---------------------------------------------------------------------------
// 6a) BN + ReLU -> next layer's f16 input (ICpad_next == OC, always mult of 32)
// ---------------------------------------------------------------------------
__global__ void bn_relu_x_kernel(const _Float16* __restrict__ Y, const float* __restrict__ stats,
                                 const float* __restrict__ gamma, const float* __restrict__ beta,
                                 size_t total, int OC, float invCount, _Float16* __restrict__ Xn) {
    size_t i = (size_t)blockIdx.x * blockDim.x + threadIdx.x;
    if (i >= total) return;
    int c = (int)(i % OC);
    float mean = stats[c] * invCount;
    float var  = stats[256 + c] * invCount - mean * mean;
    float v = gamma[c] * ((float)Y[i] - mean) * rsqrtf(var + BN_EPS) + beta[c];
    Xn[i] = (_Float16)(v > 0.f ? v : 0.f);
}

// ---------------------------------------------------------------------------
// 6b) Final layer: BN + ReLU + max over K -> fp32 feat output (b,t,640,S)
// ---------------------------------------------------------------------------
__global__ void bn_relu_max_kernel(const _Float16* __restrict__ Y, const float* __restrict__ stats,
                                   const float* __restrict__ gamma, const float* __restrict__ beta,
                                   int K, int OC, float invCount, int co_off,
                                   float* __restrict__ feat) {
    int i = blockIdx.x * blockDim.x + threadIdx.x;
    if (i >= NBATCH * NPOINT * OC) return;
    int g = i / OC, c = i % OC;
    float mean = stats[c] * invCount;
    float inv  = rsqrtf(stats[256 + c] * invCount - mean * mean + BN_EPS);
    float ga = gamma[c], be = beta[c];
    const _Float16* yb = Y + (size_t)g * K * OC + c;
    float m = -1e30f;
    for (int k = 0; k < K; ++k) {
        float v = ga * ((float)yb[(size_t)k * OC] - mean) * inv + be;
        v = v > 0.f ? v : 0.f;
        if (v > m) m = v;
    }
    int B8 = g >> 8, s = g & 255;
    feat[((size_t)B8 * 640 + co_off + c) * NPOINT + s] = m;
}

// ---------------------------------------------------------------------------
// Host-side orchestration
// ---------------------------------------------------------------------------
extern "C" void kernel_launch(void* const* d_in, const int* in_sizes, int n_in,
                              void* d_out, int out_size, void* d_ws, size_t ws_size,
                              hipStream_t stream) {
    (void)in_sizes; (void)n_in; (void)out_size; (void)ws_size;
    const float* xyz = (const float*)d_in[0];      // (2,4,3,1024)
    const float* pts = (const float*)d_in[1];      // (2,4,128,1024)
    float* out = (float*)d_out;                    // 6144 newxyz + 1310720 feat
    float* feat = out + NBATCH * 3 * NPOINT;

    char* ws = (char*)d_ws;
    int*      fps_idx = (int*)(ws + 0);                 //   8 KB
    float*    newxyz  = (float*)(ws + 8192);            //  24 KB
    float*    stats   = (float*)(ws + 32768);           //   2 KB (sum | sumsq)
    _Float16* wgt     = (_Float16*)(ws + 40960);        // <=80 KB f16 weights
    _Float16* X       = (_Float16*)(ws + 262144);       // <=40 MB f16 activations
    _Float16* Y       = (_Float16*)(ws + 42205184ull);  // <=64 MB f16 pre-BN output

    fps_kernel<<<NBATCH, 256, 0, stream>>>(xyz, fps_idx, newxyz, out);

    const int   Ks[3]      = {16, 32, 64};
    const float rad[3]     = {0.2f, 0.4f, 0.8f};
    const int   mlp[3][3]  = {{64, 64, 128}, {128, 128, 256}, {128, 128, 256}};
    const int   co_off[3]  = {0, 128, 384};

    for (int br = 0; br < 3; ++br) {
        const int K = Ks[br];
        const int P = NBATCH * NPOINT * K;          // 32768 / 65536 / 131072
        const float r2 = rad[br] * rad[br];
        group_kernel<<<NBATCH * NPOINT, 128, 0, stream>>>(xyz, pts, newxyz, r2, K, 160, X);

        int IC = 131, ICpad = 160;
        for (int l = 0; l < 3; ++l) {
            const int OC = mlp[br][l];
            const int pi = 2 + (br * 3 + l) * 4;
            const float* W     = (const float*)d_in[pi + 0];
            const float* bias  = (const float*)d_in[pi + 1];
            const float* gamma = (const float*)d_in[pi + 2];
            const float* beta  = (const float*)d_in[pi + 3];

            wconvert_kernel<<<(OC * ICpad + 255) / 256, 256, 0, stream>>>(W, wgt, OC, IC, ICpad);
            dim3 ggrid(P / 16, OC / 64);
            gemm_bias_kernel<<<ggrid, 128, 0, stream>>>(X, wgt, bias, Y, P, OC, ICpad);
            stats_kernel<<<OC, 256, 0, stream>>>(Y, P, OC, stats);
            const float invCount = 1.0f / (float)P;
            if (l < 2) {
                size_t total = (size_t)P * OC;
                bn_relu_x_kernel<<<(unsigned)((total + 255) / 256), 256, 0, stream>>>(
                    Y, stats, gamma, beta, total, OC, invCount, X);
                IC = OC; ICpad = OC;                 // 64/128: multiples of 32, no repad
            } else {
                bn_relu_max_kernel<<<(NBATCH * NPOINT * OC + 255) / 256, 256, 0, stream>>>(
                    Y, stats, gamma, beta, K, OC, invCount, co_off[br], feat);
            }
        }
    }
}
